// BRCLayer_88587995447898
// MI455X (gfx1250) — matrix operand order repaired and checked by reading, not compiled, not run on hardware
//
#include <hip/hip_runtime.h>
#include <hip/hip_bf16.h>
#include <cstddef>
#include <cstdint>

// ---------------------------------------------------------------------------
// BRC layer: SEQ=2048, BATCH=128, INPUT=256, HIDDEN=512
//   Phase 0a: U_c|U_a|U_h -> bf16 weight image Wb[1536][256]; bias[1536]
//   Phase 0b: x_seq f32 -> bf16 image xb[262144][256]
//   Phase 1 : Y[m][n] = xb[m][:] @ Wb[n][:]^T + bias[n]  via bf16 WMMA,
//             A tile DMA'd into LDS by the Tensor Data Mover (padded stride)
//   Phase 2 : per-(b,h) independent recurrence over t (elementwise scan)
// Workspace layout (bytes):
//   [0, 786432)              Wb   : 1536*256 bf16
//   [786432, 792576)         bias : 1536 f32
//   [1048576, 135266304)     xb   : 262144*256 bf16 (128 MB)
//   [135266304, ...)         Y    : 262144*1536 f32 (~1.61 GB)
// ---------------------------------------------------------------------------

#define SEQ     2048
#define BATCH   128
#define INPUT   256
#define HIDDEN  512
#define NTOT    (3 * HIDDEN)            // 1536
#define MTOT    (SEQ * BATCH)           // 262144
#define LDSK    264                     // padded LDS row stride (bf16 elems)

typedef __attribute__((ext_vector_type(8)))  __bf16   v8bf;
typedef __attribute__((ext_vector_type(16))) __bf16   v16bf;
typedef __attribute__((ext_vector_type(8)))  float    v8f;
typedef __attribute__((ext_vector_type(4)))  unsigned v4u;
typedef __attribute__((ext_vector_type(8)))  int      v8i;
typedef __attribute__((ext_vector_type(4)))  int      v4i;

static __device__ __forceinline__ __bf16 f2bf(float f) {
    union { float f; unsigned u; } v; v.f = f;
    unsigned u = v.u;
    unsigned r = (u + 0x7FFFu + ((u >> 16) & 1u)) >> 16;   // round-to-nearest-even
    unsigned short s = (unsigned short)r;
    __bf16 out;
    __builtin_memcpy(&out, &s, 2);
    return out;
}

static __device__ __forceinline__ v16bf cat16(v8bf lo, v8bf hi) {
    return __builtin_shufflevector(lo, hi, 0,1,2,3,4,5,6,7,8,9,10,11,12,13,14,15);
}

static __device__ __forceinline__ v8f bcast8(float s) {
    v8f r;
#pragma unroll
    for (int i = 0; i < 8; ++i) r[i] = s;
    return r;
}

// stable fast activations built on v_exp_f32
static __device__ __forceinline__ float sigmoid_f(float x) {
    return 1.0f / (1.0f + __expf(-x));
}
static __device__ __forceinline__ float tanh_f(float x) {
    return 1.0f - 2.0f / (__expf(2.0f * x) + 1.0f);
}

// ---------------------------------------------------------------------------
// Phase 0a: weight conversion + bias concat
// ---------------------------------------------------------------------------
__global__ __launch_bounds__(256) void brc_prep_w(
    const float* __restrict__ Uc, const float* __restrict__ Ua,
    const float* __restrict__ Uh, const float* __restrict__ bc,
    const float* __restrict__ ba, __bf16* __restrict__ Wb,
    float* __restrict__ bias)
{
    int i = blockIdx.x * 256 + threadIdx.x;    // one thread per 8 elements
    if (i < (NTOT * INPUT) / 8) {
        int n = i >> 5;                         // row (0..1535), 32 groups/row
        int g = i & 31;                         // 8-elem group within row
        const float* U = (n < HIDDEN) ? Uc : ((n < 2 * HIDDEN) ? Ua : Uh);
        const float4* src = (const float4*)(U + (size_t)(n & (HIDDEN - 1)) * INPUT + g * 8);
        float4 a = src[0], b = src[1];
        v8bf o = { f2bf(a.x), f2bf(a.y), f2bf(a.z), f2bf(a.w),
                   f2bf(b.x), f2bf(b.y), f2bf(b.z), f2bf(b.w) };
        *(v8bf*)(Wb + (size_t)i * 8) = o;       // 16 B store
    }
    if (i < NTOT) {
        bias[i] = (i < HIDDEN) ? bc[i]
                : (i < 2 * HIDDEN) ? ba[i - HIDDEN] : 0.0f;
    }
}

// ---------------------------------------------------------------------------
// Phase 0b: x_seq f32 -> bf16 image (streaming, packed 16 B stores)
// ---------------------------------------------------------------------------
__global__ __launch_bounds__(256) void brc_prep_x(
    const float* __restrict__ x, __bf16* __restrict__ xb)
{
    size_t i = (size_t)blockIdx.x * 256 + threadIdx.x;   // one thread per 8 elems
    const float4* src = (const float4*)(x + i * 8);
    float4 a = src[0], b = src[1];
    v8bf o = { f2bf(a.x), f2bf(a.y), f2bf(a.z), f2bf(a.w),
               f2bf(b.x), f2bf(b.y), f2bf(b.z), f2bf(b.w) };
    *(v8bf*)(xb + i * 8) = o;
}

// ---------------------------------------------------------------------------
// Phase 1: Y = xb @ Wb^T + bias     (M=262144, N=1536, K=256)
// Block: 256 threads = 8 wave32s; tile M=64 x N=64.
//   wave (w&3) -> 16-row M sub-tile, (w>>2) -> 32-col N sub-tile (2 accums)
// A tile (64 x 256 bf16) DMA'd into LDS by the TDM with pad_enable giving a
// 264-element row stride (kills ds bank conflicts). B fragments stream from
// the 768 KB L2-resident weight image.
// ---------------------------------------------------------------------------
__global__ __launch_bounds__(256) void brc_gemm(
    const __bf16* __restrict__ xb, const __bf16* __restrict__ Wb,
    const float* __restrict__ bias, float* __restrict__ Y)
{
    extern __shared__ __bf16 As[];                // 64 x LDSK bf16 = 33792 B

    const int tid  = threadIdx.x;
    const int lane = tid & 31;
    const int w    = tid >> 5;
    const int m0   = blockIdx.x * 64;
    const int n0   = blockIdx.y * 64;

    // ---- TDM: DMA A tile (64 rows x 256 bf16) global -> LDS, padded ----
    if (w == 0) {
        uint64_t gaddr = (uint64_t)(const void*)(xb + (size_t)m0 * INPUT);
        unsigned ldsa  = (unsigned)(size_t)(void*)As;   // LDS byte offset

        // D# group 0: count=1 | lds_addr | global_addr[56:0] | type=2
        v4u g0 = { 1u,
                   ldsa,
                   (unsigned)(gaddr & 0xFFFFFFFFu),
                   (unsigned)((gaddr >> 32) & 0x01FFFFFFu) | (2u << 30) };

        // D# group 1: data_size=2B(1), pad_enable, pad_interval=128dw(6),
        //             pad_amount=4dw(3); tensor_dim0=256, tensor_dim1=262144,
        //             tile_dim0=256, tile_dim1=64, tensor_dim0_stride=256
        v8i g1 = { (int)((1u << 16) | (1u << 20) | (6u << 22) | (3u << 25)),
                   (int)(256u << 16),            // tensor_dim0 @ bit48
                   0,                            // dim0 hi, dim1 lo16 (=0)
                   (int)(4u | (256u << 16)),     // dim1 hi16=4, tile_dim0=256
                   64,                           // tile_dim1=64, tile_dim2=0
                   256,                          // tensor_dim0_stride lo32
                   0, 0 };
        v4i gz4 = { 0, 0, 0, 0 };                // groups 2/3: 2-D tensor
        v8i gz8 = { 0, 0, 0, 0, 0, 0, 0, 0 };

        __builtin_amdgcn_tensor_load_to_lds(g0, g1, gz4, gz4, gz8, 0);
        __builtin_amdgcn_s_wait_tensorcnt(0);
    }
    __syncthreads();

    const int l16   = lane & 15;
    const int halfk = lane >> 4;                  // k-half select (ISA A/B layout)
    const int mrow  = (w & 3) * 16 + l16;         // A row in LDS for this lane
    const int nsub  = (w >> 2) * 32;              // 0 or 32
    const int ncol0 = n0 + nsub + l16;            // j=0 column
    const int ncol1 = ncol0 + 16;                 // j=1 column

    v8f acc0 = bcast8(bias[ncol0]);               // fold bias into C
    v8f acc1 = bcast8(bias[ncol1]);

#pragma unroll
    for (int kc = 0; kc < 8; ++kc) {
        const int k0 = kc * 32;

        // A fragment 16x32: lane holds row mrow, k in [k0+8h, k0+8h+8) and +16
        const __bf16* ap = &As[mrow * LDSK + k0 + halfk * 8];
        v16bf afrag = cat16(*(const v8bf*)ap, *(const v8bf*)(ap + 16));

        // B fragments 32x16: lane holds column, k in [k0+8h, k0+8h+8) and +16
        const __bf16* bp0 = Wb + (size_t)ncol0 * INPUT + k0 + halfk * 8;
        v16bf bfrag0 = cat16(*(const v8bf*)bp0, *(const v8bf*)(bp0 + 16));
        const __bf16* bp1 = bp0 + 16 * INPUT;
        v16bf bfrag1 = cat16(*(const v8bf*)bp1, *(const v8bf*)(bp1 + 16));

        acc0 = __builtin_amdgcn_wmma_f32_16x16x32_bf16(
                   false, afrag, false, bfrag0, (short)0, acc0, false, false);
        acc1 = __builtin_amdgcn_wmma_f32_16x16x32_bf16(
                   false, afrag, false, bfrag1, (short)0, acc1, false, false);
    }

    // D layout: VGPR v, lanes0-15 -> M=v, lanes16-31 -> M=v+8
    const size_t rowbase = (size_t)m0 + (w & 3) * 16 + halfk * 8;
#pragma unroll
    for (int v = 0; v < 8; ++v) {
        size_t row = rowbase + v;
        Y[row * NTOT + ncol0] = acc0[v];
        Y[row * NTOT + ncol1] = acc1[v];
    }
}

// ---------------------------------------------------------------------------
// Phase 2: elementwise scan. One lane per (b,h) element; 65536 lanes total.
//   c = sigmoid(xc + w_c*h); a = 1 + tanh(xa + w_a*h)
//   h' = c*h + (1-c)*tanh(xh + a*h)
// ---------------------------------------------------------------------------
__global__ __launch_bounds__(256) void brc_scan(
    const float* __restrict__ Y, const float* __restrict__ h0,
    const float* __restrict__ wc_, const float* __restrict__ wa_,
    float* __restrict__ out)
{
    const int g = blockIdx.x * 256 + threadIdx.x;   // < 65536
    const int b = g >> 9;
    const int h = g & (HIDDEN - 1);

    const float wc = wc_[h];
    const float wa = wa_[h];
    float hv = h0[g];

    const float* yb = Y + (size_t)b * NTOT + h;
    const size_t tstride = (size_t)BATCH * NTOT;    // 196608 floats per t

    for (int t = 0; t < SEQ; ++t) {
        size_t o = (size_t)t * tstride;
        __builtin_prefetch(yb + o + 4 * tstride, 0, 0);   // global_prefetch_b8
        float xc = yb[o];
        float xa = yb[o + HIDDEN];
        float xh = yb[o + 2 * HIDDEN];

        float c  = sigmoid_f(xc + wc * hv);
        float a  = 2.0f - 2.0f / (__expf(2.0f * (xa + wa * hv)) + 1.0f); // 1+tanh
        float hh = tanh_f(xh + a * hv);
        hv = c * hv + (1.0f - c) * hh;

        out[(size_t)t * (BATCH * HIDDEN) + g] = hv;
    }
    // final state hn appended after the (S,B,H) trajectory
    out[(size_t)SEQ * (BATCH * HIDDEN) + g] = hv;
}

// ---------------------------------------------------------------------------
// Host launcher
// d_in order: x_seq, h, U_c, w_c, b_c, U_a, w_a, b_a, U_h
// ---------------------------------------------------------------------------
extern "C" void kernel_launch(void* const* d_in, const int* in_sizes, int n_in,
                              void* d_out, int out_size, void* d_ws, size_t ws_size,
                              hipStream_t stream) {
    const float* x   = (const float*)d_in[0];
    const float* h0  = (const float*)d_in[1];
    const float* Uc  = (const float*)d_in[2];
    const float* wc  = (const float*)d_in[3];
    const float* bc  = (const float*)d_in[4];
    const float* Ua  = (const float*)d_in[5];
    const float* wa  = (const float*)d_in[6];
    const float* ba  = (const float*)d_in[7];
    const float* Uh  = (const float*)d_in[8];
    float* out = (float*)d_out;

    char* ws = (char*)d_ws;
    __bf16* Wb   = (__bf16*)(ws);                       // 786432 B
    float*  bias = (float*)(ws + 786432);               // 6144 B
    __bf16* xb   = (__bf16*)(ws + 1048576);             // 128 MB
    float*  Y    = (float*)(ws + 1048576 + (size_t)MTOT * INPUT * 2);

    // Phase 0a: weights -> bf16, bias concat
    brc_prep_w<<<((NTOT * INPUT) / 8 + 255) / 256, 256, 0, stream>>>(
        Uc, Ua, Uh, bc, ba, Wb, bias);

    // Phase 0b: x -> bf16 image (67,108,864 elems, 8 per thread)
    brc_prep_x<<<(MTOT * INPUT / 8) / 256, 256, 0, stream>>>(x, xb);

    // Phase 1: fused projection GEMM (M=262144, N=1536, K=256)
    dim3 ggrid(MTOT / 64, NTOT / 64);
    brc_gemm<<<ggrid, 256, 64 * LDSK * sizeof(__bf16), stream>>>(xb, Wb, bias, Y);

    // Phase 2: recurrence scan, one lane per (b,h)
    brc_scan<<<(BATCH * HIDDEN) / 256, 256, 0, stream>>>(Y, h0, wc, wa, out);
}
